// FullModel_49976239456889
// MI455X (gfx1250) — compile-verified, matrix operand order checked by reading, not run on hardware
//
#include <hip/hip_runtime.h>
#include <math.h>

// ---------------------------------------------------------------------------
// Types for CDNA5 WMMA
// ---------------------------------------------------------------------------
typedef __attribute__((ext_vector_type(16))) __bf16 v16bf;
typedef __attribute__((ext_vector_type(8)))  __bf16 v8bf;
typedef __attribute__((ext_vector_type(4)))  __bf16 v4bf;
typedef __attribute__((ext_vector_type(8)))  float  v8f;

#define ATOMIC_ADD_F32(p, v) \
    __hip_atomic_fetch_add((p), (v), __ATOMIC_RELAXED, __HIP_MEMORY_SCOPE_AGENT)

// ---------------------------------------------------------------------------
// Utility kernels
// ---------------------------------------------------------------------------
__global__ void fill_kernel(float* __restrict__ p, float v, size_t n) {
    size_t i = (size_t)blockIdx.x * blockDim.x + threadIdx.x;
    if (i < n) p[i] = v;
}

// L2-normalize rows of x (D == 128). One wave (32 lanes) per row, float4/lane.
__global__ __launch_bounds__(256) void l2norm_kernel(
    const float* __restrict__ x, float* __restrict__ out, int n) {
    int wave = threadIdx.x >> 5;
    int lane = threadIdx.x & 31;
    int row  = blockIdx.x * 8 + wave;
    if (row >= n) return;
    const float4 v = ((const float4*)(x + (size_t)row * 128))[lane];
    float s = v.x * v.x + v.y * v.y + v.z * v.z + v.w * v.w;
#pragma unroll
    for (int off = 16; off > 0; off >>= 1) s += __shfl_xor(s, off, 32);
    float inv = 1.0f / fmaxf(sqrtf(s), 1e-12f);
    float4 o;
    o.x = v.x * inv; o.y = v.y * inv; o.z = v.z * inv; o.w = v.w * inv;
    ((float4*)(out + (size_t)row * 128))[lane] = o;
}

// Per-destination edge counts (as float, for the mean divide).
__global__ void count_kernel(const int* __restrict__ dst,
                             float* __restrict__ cnt, int E) {
    int e = blockIdx.x * blockDim.x + threadIdx.x;
    if (e < E) ATOMIC_ADD_F32(&cnt[dst[e]], 1.0f);
}

// agg[dst[e], :] += feat[src[e], :]   (D multiple of 4)
__global__ void scatter_add_kernel(const float* __restrict__ feat,
                                   const int* __restrict__ src,
                                   const int* __restrict__ dst,
                                   float* __restrict__ agg, int E, int D) {
    size_t t     = (size_t)blockIdx.x * blockDim.x + threadIdx.x;
    int    per   = D >> 2;
    size_t total = (size_t)E * per;
    if (t >= total) return;
    int e = (int)(t / per);
    int c = (int)(t % per) << 2;
    int s = src[e];
    int d = dst[e];
    const float4 v = *(const float4*)(feat + (size_t)s * D + c);
    float* p = agg + (size_t)d * D + c;
    ATOMIC_ADD_F32(p + 0, v.x);
    ATOMIC_ADD_F32(p + 1, v.y);
    ATOMIC_ADD_F32(p + 2, v.z);
    ATOMIC_ADD_F32(p + 3, v.w);
}

// agg[i, :] /= max(cnt[i], 1)
__global__ void mean_div_kernel(float* __restrict__ agg,
                                const float* __restrict__ cnt,
                                int n, int D) {
    size_t t = (size_t)blockIdx.x * blockDim.x + threadIdx.x;
    if (t >= (size_t)n * D) return;
    int i = (int)(t / D);
    agg[t] = agg[t] / fmaxf(cnt[i], 1.0f);
}

// ---------------------------------------------------------------------------
// WMMA GEMM:  C[M,N] = act( (ADD ? C : 0) + A[M,K] * W[N,K]^T + (BIAS ? b : 0) )
//
// A, W fp32 row-major; converted to bf16 during LDS staging; fp32 accumulation
// via v_wmma_f32_16x16x32_bf16.
//   Block tile : 128(M) x 128(N), K-step 32, double-buffered LDS.
//   Waves      : 8 = 4(M) x 2(N); each wave owns 32x64 -> 8 acc tiles,
//                8 WMMA per k-step (B fragment reused for 2 A fragments).
//   LDS rows   : LDK=40 bf16 (80B stride) -> 16B-aligned fragment chunks
//                (ds_load_b128) with conflict-free bank spread (20r mod 64).
// Requires K % 32 == 0 (true here: K is 128 or 256).
// ---------------------------------------------------------------------------
#define BM  128
#define BN  128
#define BK  32
#define LDK 40

template <int ACT, bool ADD, bool BIAS>
__global__ __launch_bounds__(256) void gemm_bf16_wmma(
    const float* __restrict__ A, const float* __restrict__ W,
    const float* __restrict__ bias, float* __restrict__ C,
    int M, int N, int K, int ldc) {
    __shared__ __align__(16) __bf16 As[2][BM * LDK];
    __shared__ __align__(16) __bf16 Bs[2][BN * LDK];

    const int tid  = threadIdx.x;
    const int wave = tid >> 5;
    const int lane = tid & 31;
    const int g    = lane >> 4;    // lane group (K-half select)
    const int r    = lane & 15;    // row/col within fragment
    const int wm   = wave & 3;     // wave M position (0..3) -> rows wm*32
    const int wn   = wave >> 2;    // wave N position (0..1) -> cols wn*64

    const int block_m = blockIdx.y * BM;
    const int block_n = blockIdx.x * BN;

    const int trow = tid >> 3;        // 0..31 (tile row per load iter)
    const int tcol = (tid & 7) << 2;  // 0,4,...,28 (float4 column)

    const bool a_full = (block_m + BM) <= M;
    const bool b_full = (block_n + BN) <= N;

    // Stage one 128x32 K-slice of A and W into LDS buffer `buf` as bf16.
    auto stage = [&](int buf, int k0) {
#pragma unroll
        for (int it = 0; it < 4; ++it) {
            int row = trow + it * 32;
            int gm  = block_m + row;
            float4 v = {0.f, 0.f, 0.f, 0.f};
            if (a_full || gm < M) v = *(const float4*)(A + (size_t)gm * K + k0 + tcol);
            v4bf pk;
            pk[0] = (__bf16)v.x; pk[1] = (__bf16)v.y;
            pk[2] = (__bf16)v.z; pk[3] = (__bf16)v.w;
            *(v4bf*)&As[buf][row * LDK + tcol] = pk;
        }
#pragma unroll
        for (int it = 0; it < 4; ++it) {
            int row = trow + it * 32;
            int gn  = block_n + row;
            float4 v = {0.f, 0.f, 0.f, 0.f};
            if (b_full || gn < N) v = *(const float4*)(W + (size_t)gn * K + k0 + tcol);
            v4bf pk;
            pk[0] = (__bf16)v.x; pk[1] = (__bf16)v.y;
            pk[2] = (__bf16)v.z; pk[3] = (__bf16)v.w;
            *(v4bf*)&Bs[buf][row * LDK + tcol] = pk;
        }
    };

    v8f acc[2][4];
    const v8f vzero = {0.f, 0.f, 0.f, 0.f, 0.f, 0.f, 0.f, 0.f};
#pragma unroll
    for (int a = 0; a < 2; ++a)
#pragma unroll
        for (int nt = 0; nt < 4; ++nt) acc[a][nt] = vzero;

    stage(0, 0);
    __syncthreads();

    for (int k0 = 0; k0 < K; k0 += BK) {
        const int cur = (k0 / BK) & 1;
        if (k0 + BK < K) stage(cur ^ 1, k0 + BK);  // prefetch next slice

        // A fragments: rows wm*32 + a*16 + r; K elems {8g..8g+7, 16+8g..23+8g}
        v16bf af[2];
#pragma unroll
        for (int a = 0; a < 2; ++a) {
            const int arow = wm * 32 + a * 16 + r;
            v8bf lo = *(const v8bf*)&As[cur][arow * LDK + (g << 3)];
            v8bf hi = *(const v8bf*)&As[cur][arow * LDK + 16 + (g << 3)];
            af[a] = __builtin_shufflevector(lo, hi, 0, 1, 2, 3, 4, 5, 6, 7,
                                            8, 9, 10, 11, 12, 13, 14, 15);
        }
#pragma unroll
        for (int nt = 0; nt < 4; ++nt) {
            const int brow = wn * 64 + nt * 16 + r;
            v8bf lo = *(const v8bf*)&Bs[cur][brow * LDK + (g << 3)];
            v8bf hi = *(const v8bf*)&Bs[cur][brow * LDK + 16 + (g << 3)];
            v16bf bf = __builtin_shufflevector(lo, hi, 0, 1, 2, 3, 4, 5, 6, 7,
                                               8, 9, 10, 11, 12, 13, 14, 15);
            acc[0][nt] = __builtin_amdgcn_wmma_f32_16x16x32_bf16(
                false, af[0], false, bf, (short)0, acc[0][nt], false, false);
            acc[1][nt] = __builtin_amdgcn_wmma_f32_16x16x32_bf16(
                false, af[1], false, bf, (short)0, acc[1][nt], false, false);
        }
        __syncthreads();
    }

    // Epilogue: D layout -> element (m = j + 8*g, n = r) in acc[a][nt][j]
#pragma unroll
    for (int a = 0; a < 2; ++a) {
#pragma unroll
        for (int nt = 0; nt < 4; ++nt) {
#pragma unroll
            for (int j = 0; j < 8; ++j) {
                int m = block_m + wm * 32 + a * 16 + j + (g << 3);
                int n = block_n + wn * 64 + nt * 16 + r;
                if (m < M && n < N) {
                    float v = acc[a][nt][j];
                    if (BIAS) v += bias[n];
                    if (ADD)  v += C[(size_t)m * ldc + n];
                    if (ACT == 1) v = fmaxf(v, 0.0f);
                    if (ACT == 2) v = 1.0f / (1.0f + __expf(-v));
                    C[(size_t)m * ldc + n] = v;
                }
            }
        }
    }
}

// ---------------------------------------------------------------------------
// Host-side launch
// ---------------------------------------------------------------------------
static inline unsigned cdiv(size_t a, size_t b) { return (unsigned)((a + b - 1) / b); }

template <int ACT, bool ADD, bool BIAS>
static void launch_gemm(const float* A, const float* W, const float* bias,
                        float* C, int M, int N, int K, int ldc,
                        hipStream_t stream) {
    dim3 grid(cdiv(N, BN), cdiv(M, BM));
    gemm_bf16_wmma<ACT, ADD, BIAS><<<grid, 256, 0, stream>>>(A, W, bias, C, M, N, K, ldc);
}

extern "C" void kernel_launch(void* const* d_in, const int* in_sizes, int n_in,
                              void* d_out, int out_size, void* d_ws, size_t ws_size,
                              hipStream_t stream) {
    // ---- inputs (setup_inputs order) ----
    const float* x_client  = (const float*)d_in[0];
    const float* x_sku     = (const float*)d_in[1];
    const int*   ei_cs_src = (const int*)d_in[2];
    const int*   ei_cs_dst = (const int*)d_in[3];
    const int*   ei_sc_src = (const int*)d_in[4];
    const int*   ei_sc_dst = (const int*)d_in[5];
    const float* W1l_cs = (const float*)d_in[6];
    const float* b1_cs  = (const float*)d_in[7];
    const float* W1r_cs = (const float*)d_in[8];
    const float* W1l_sc = (const float*)d_in[9];
    const float* b1_sc  = (const float*)d_in[10];
    const float* W1r_sc = (const float*)d_in[11];
    const float* Wlin1_c = (const float*)d_in[12];
    const float* blin1_c = (const float*)d_in[13];
    const float* Wlin1_s = (const float*)d_in[14];
    const float* blin1_s = (const float*)d_in[15];
    const float* W2l_sc  = (const float*)d_in[19];
    const float* b2_sc   = (const float*)d_in[20];
    const float* W2r_sc  = (const float*)d_in[21];
    const float* Wlin2_c = (const float*)d_in[22];
    const float* blin2_c = (const float*)d_in[23];
    const float* Wch1 = (const float*)d_in[26];
    const float* bch1 = (const float*)d_in[27];
    const float* Wch2 = (const float*)d_in[28];
    const float* bch2 = (const float*)d_in[29];
    const float* Wcat1 = (const float*)d_in[30];
    const float* bcat1 = (const float*)d_in[31];
    const float* Wcat2 = (const float*)d_in[32];
    const float* bcat2 = (const float*)d_in[33];
    const float* Wsk1 = (const float*)d_in[34];
    const float* bsk1 = (const float*)d_in[35];
    const float* Wsk2 = (const float*)d_in[36];
    const float* bsk2 = (const float*)d_in[37];

    const int H    = in_sizes[7];            // 256
    const int D    = in_sizes[6] / H;        // 128
    const int Nc   = in_sizes[0] / D;        // 100000
    const int Ns   = in_sizes[1] / D;        // 20000
    const int E1   = in_sizes[2];            // cs edges
    const int E2   = in_sizes[4];            // sc edges
    const int O    = in_sizes[20];           // 128 (b2_sc)
    const int HH   = in_sizes[27];           // 128 head hidden
    const int NCAT = in_sizes[33];           // 200
    const int NSKU = in_sizes[37];           // 1000

    // ---- workspace layout (floats) ----
    float* ws = (float*)d_ws;
    size_t o = 0;
    float* xc_n  = ws + o; o += (size_t)Nc * D;
    float* xs_n  = ws + o; o += (size_t)Ns * D;
    float* agg_s = ws + o; o += (size_t)Ns * D;
    float* agg_c = ws + o; o += (size_t)Nc * D;   // later reused as head hidden
    float* cnt_s = ws + o; o += (size_t)Ns;
    float* cnt_c = ws + o; o += (size_t)Nc;
    float* h1_s  = ws + o; o += (size_t)Ns * H;
    float* h1_c  = ws + o; o += (size_t)Nc * H;
    float* agg2  = ws + o; o += (size_t)Nc * H;
    float* hidden = agg_c;                        // alias (free after layer-1)

    float* out       = (float*)d_out;
    float* out_churn = out;
    float* out_cat   = out + (size_t)Nc;
    float* out_sku   = out + (size_t)Nc * (1 + NCAT);
    float* user_emb  = out + (size_t)Nc * (1 + NCAT + NSKU);

    const int T = 256;

    // ---- zero accumulators ----
    {
        size_t n;
        n = (size_t)Ns * D; fill_kernel<<<cdiv(n, T), T, 0, stream>>>(agg_s, 0.f, n);
        n = (size_t)Nc * D; fill_kernel<<<cdiv(n, T), T, 0, stream>>>(agg_c, 0.f, n);
        n = (size_t)Ns;     fill_kernel<<<cdiv(n, T), T, 0, stream>>>(cnt_s, 0.f, n);
        n = (size_t)Nc;     fill_kernel<<<cdiv(n, T), T, 0, stream>>>(cnt_c, 0.f, n);
        n = (size_t)Nc * H; fill_kernel<<<cdiv(n, T), T, 0, stream>>>(agg2, 0.f, n);
    }

    // ---- l2 normalize inputs ----
    l2norm_kernel<<<cdiv(Nc, 8), T, 0, stream>>>(x_client, xc_n, Nc);
    l2norm_kernel<<<cdiv(Ns, 8), T, 0, stream>>>(x_sku, xs_n, Ns);

    // ---- edge counts ----
    count_kernel<<<cdiv(E1, T), T, 0, stream>>>(ei_cs_dst, cnt_s, E1);
    count_kernel<<<cdiv(E2, T), T, 0, stream>>>(ei_sc_dst, cnt_c, E2);

    // ---- layer-1 aggregations (mean of normalized features) ----
    scatter_add_kernel<<<cdiv((size_t)E1 * (D >> 2), T), T, 0, stream>>>(
        xc_n, ei_cs_src, ei_cs_dst, agg_s, E1, D);
    scatter_add_kernel<<<cdiv((size_t)E2 * (D >> 2), T), T, 0, stream>>>(
        xs_n, ei_sc_src, ei_sc_dst, agg_c, E2, D);
    mean_div_kernel<<<cdiv((size_t)Ns * D, T), T, 0, stream>>>(agg_s, cnt_s, Ns, D);
    mean_div_kernel<<<cdiv((size_t)Nc * D, T), T, 0, stream>>>(agg_c, cnt_c, Nc, D);

    // ---- layer-1 GEMMs (bf16 WMMA, fp32 accum) ----
    // h1_s = relu(mean_s@W1l_cs^T + b1_cs + xs_n@W1r_cs^T + xs_n@Wlin1_s^T + blin1_s)
    launch_gemm<0, false, true>(agg_s, W1l_cs, b1_cs,  h1_s, Ns, H, D, H, stream);
    launch_gemm<0, true, false>(xs_n,  W1r_cs, nullptr, h1_s, Ns, H, D, H, stream);
    launch_gemm<1, true, true>( xs_n,  Wlin1_s, blin1_s, h1_s, Ns, H, D, H, stream);
    // h1_c = relu(mean_c@W1l_sc^T + b1_sc + xc_n@W1r_sc^T + xc_n@Wlin1_c^T + blin1_c)
    launch_gemm<0, false, true>(agg_c, W1l_sc, b1_sc,  h1_c, Nc, H, D, H, stream);
    launch_gemm<0, true, false>(xc_n,  W1r_sc, nullptr, h1_c, Nc, H, D, H, stream);
    launch_gemm<1, true, true>( xc_n,  Wlin1_c, blin1_c, h1_c, Nc, H, D, H, stream);

    // ---- layer-2 aggregation: mean over sc edges of h1_s at clients ----
    scatter_add_kernel<<<cdiv((size_t)E2 * (H >> 2), T), T, 0, stream>>>(
        h1_s, ei_sc_src, ei_sc_dst, agg2, E2, H);
    mean_div_kernel<<<cdiv((size_t)Nc * H, T), T, 0, stream>>>(agg2, cnt_c, Nc, H);

    // ---- layer-2 GEMMs -> user_emb (also head input) ----
    launch_gemm<0, false, true>(agg2, W2l_sc, b2_sc,   user_emb, Nc, O, H, O, stream);
    launch_gemm<0, true, false>(h1_c, W2r_sc, nullptr, user_emb, Nc, O, H, O, stream);
    launch_gemm<0, true, true>( h1_c, Wlin2_c, blin2_c, user_emb, Nc, O, H, O, stream);

    // ---- heads: sigmoid(relu(u@W1^T+b1)@W2^T+b2) ----
    launch_gemm<1, false, true>(user_emb, Wch1, bch1, hidden, Nc, HH, O, HH, stream);
    launch_gemm<2, false, true>(hidden, Wch2, bch2, out_churn, Nc, 1, HH, 1, stream);

    launch_gemm<1, false, true>(user_emb, Wcat1, bcat1, hidden, Nc, HH, O, HH, stream);
    launch_gemm<2, false, true>(hidden, Wcat2, bcat2, out_cat, Nc, NCAT, HH, NCAT, stream);

    launch_gemm<1, false, true>(user_emb, Wsk1, bsk1, hidden, Nc, HH, O, HH, stream);
    launch_gemm<2, false, true>(hidden, Wsk2, bsk2, out_sku, Nc, NSKU, HH, NSKU, stream);
}
